// MutualAttention_register_76527727280231
// MI455X (gfx1250) — matrix operand order repaired and checked by reading, not compiled, not run on hardware
//
#include <hip/hip_runtime.h>
#include <hip/hip_bf16.h>
#include <math.h>

// ---------------- types ----------------
typedef __bf16 v16bf __attribute__((ext_vector_type(16)));
typedef __bf16 v8bf  __attribute__((ext_vector_type(8)));
typedef float  v8f   __attribute__((ext_vector_type(8)));
typedef float  v4f   __attribute__((ext_vector_type(4)));

#define WMMA_BF16(a, b, c) \
  __builtin_amdgcn_wmma_f32_16x16x32_bf16(false, (a), false, (b), (short)0, (c), false, false)

// ---------------- sizes ----------------
static constexpr int    Bn = 8, Cc = 64, Hh = 128, Ww = 128;
static constexpr int    Kk = Hh * Ww;              // 16384
static constexpr size_t FEAT = (size_t)Bn * Cc * Kk; // 8388608

// output offsets (floats, concatenated return order)
static constexpr size_t O1  = 0;                 // feat1_apply
static constexpr size_t O2  = 8388608;           // feat2_apply
static constexpr size_t O3  = 16777216;          // grid1 [8,128,128,2]
static constexpr size_t O4  = 17039360;          // grid2
static constexpr size_t O5  = 17301504;          // feat1_edge
static constexpr size_t O6  = 25690112;          // feat2_color
static constexpr size_t O7  = 34078720;          // f1_attn [8,64,64]
static constexpr size_t O8  = 34111488;          // f2_attn
static constexpr size_t O9  = 34144256;          // feat1 copy
static constexpr size_t O10 = 42532864;          // feat2 copy

// workspace offsets (bytes)
static constexpr size_t WS_C1   = 0;          // c1  [8,128,128] f32
static constexpr size_t WS_C2   = 524288;     // c2
static constexpr size_t WS_VSUM = 1048576;    // [8,97,128] f32
static constexpr size_t WS_SCAL = 1445888;    // x1[8] y1[8] x2[8] y2[8]
static constexpr size_t WS_FSUM = 1446144;    // sum-ch f_g [8,256,256]
static constexpr size_t WS_E256 = 3543296;    // edge @256 [8,256,256]
static constexpr size_t WS_E128 = 5640448;    // edge @128 [8,128,128]
static constexpr size_t WS_MOM  = 6164736;    // moments @256 [8,9,256,256]
static constexpr size_t WS_COL  = 25039104;   // color @128 [8,9,128,128]
static constexpr size_t WS_F2M  = 29757696;   // feat2_move bf16 [8,64,16384]
static constexpr size_t WS_F1M  = 46534912;   // feat1_edge_move bf16
static constexpr size_t WS_PART = 63312128;   // score partials [8][16][64][64] f32
static constexpr size_t WS_ATTN = 65409280;   // attn bf16 [16][64][64]
static constexpr size_t WS_BN   = 65540352;   // [2][64][2] mean,rstd

// ---------------- K1: cam upsample 64 -> 128 (half-pixel bilinear) ----------------
__global__ void ma_up2x_kernel(const float* __restrict__ src, float* __restrict__ dst) {
  int i = blockIdx.x * 256 + threadIdx.x;
  if (i >= Bn * Kk) return;
  int b = i >> 14, p = i & (Kk - 1);
  int oy = p >> 7, ox = p & 127;
  const float* s = src + (size_t)b * 64 * 64;
  float sy = (oy + 0.5f) * 0.5f - 0.5f, sx = (ox + 0.5f) * 0.5f - 0.5f;
  float y0f = floorf(sy), x0f = floorf(sx);
  float fy = sy - y0f, fx = sx - x0f;
  int y0 = min(max((int)y0f, 0), 63), y1 = min(max((int)y0f + 1, 0), 63);
  int x0 = min(max((int)x0f, 0), 63), x1 = min(max((int)x0f + 1, 0), 63);
  float v = s[y0 * 64 + x0] * (1.f - fy) * (1.f - fx) + s[y0 * 64 + x1] * (1.f - fy) * fx
          + s[y1 * 64 + x0] * fy * (1.f - fx)        + s[y1 * 64 + x1] * fy * fx;
  dst[i] = v;
}

// ---------------- K2: c1 flat-gradient centroid + first-argmax ----------------
__global__ void ma_stats1_kernel(const float* __restrict__ c1, float* __restrict__ scal) {
  const int b = blockIdx.x;
  const float* img = c1 + (size_t)b * Kk;
  float cnt = 0.f, sr = 0.f, sc = 0.f, mx = -1e30f;
  int mi = 0;
  for (int p = threadIdx.x; p < Kk; p += 256) {
    int i = p >> 7, j = p & 127;
    float v = img[p];
    float gx = (i == 0)   ? img[128 + j] - v
             : (i == 127) ? v - img[126 * 128 + j]
                          : (img[(i + 1) * 128 + j] - img[(i - 1) * 128 + j]) * 0.5f;
    float gy = (j == 0)   ? img[i * 128 + 1] - v
             : (j == 127) ? v - img[i * 128 + 126]
                          : (img[i * 128 + j + 1] - img[i * 128 + j - 1]) * 0.5f;
    if (sqrtf(gx * gx + gy * gy) < 0.1f) { cnt += 1.f; sr += (float)i; sc += (float)j; }
    if (v > mx) { mx = v; mi = p; }
  }
  __shared__ float sC[256], sR[256], sS[256], sM[256];
  __shared__ int sI[256];
  int t = threadIdx.x;
  sC[t] = cnt; sR[t] = sr; sS[t] = sc; sM[t] = mx; sI[t] = mi;
  __syncthreads();
  for (int off = 128; off > 0; off >>= 1) {
    if (t < off) {
      sC[t] += sC[t + off]; sR[t] += sR[t + off]; sS[t] += sS[t + off];
      if (sM[t + off] > sM[t] || (sM[t + off] == sM[t] && sI[t + off] < sI[t])) {
        sM[t] = sM[t + off]; sI[t] = sI[t + off];
      }
    }
    __syncthreads();
  }
  if (t == 0) {
    float x1, y1;
    if (sC[0] > 0.f) {
      x1 = floorf(sR[0] / fmaxf(sC[0], 1.f));
      y1 = floorf(sS[0] / fmaxf(sC[0], 1.f));
    } else { x1 = (float)(sI[0] >> 7); y1 = (float)(sI[0] & 127); }
    scal[b] = x1; scal[8 + b] = y1;
  }
}

// ---------------- K3a: vertical 32-row sums of c2 ----------------
__global__ void ma_vsum_kernel(const float* __restrict__ c2, float* __restrict__ vsum) {
  int i = blockIdx.x * 256 + threadIdx.x;
  if (i >= Bn * 97 * 128) return;
  int b = i / (97 * 128), rem = i % (97 * 128);
  int r0 = rem / 128, c = rem % 128;
  const float* img = c2 + (size_t)b * Kk;
  float s = 0.f;
  for (int r = 0; r < 32; ++r) s += img[(r0 + r) * 128 + c];
  vsum[i] = s;
}

// ---------------- K3b: best 32x32 window (first max) ----------------
__global__ void ma_stats2_kernel(const float* __restrict__ vsum, float* __restrict__ scal) {
  const int b = blockIdx.x;
  const float* vs = vsum + (size_t)b * 97 * 128;
  float mx = -1e30f; int mi = 0;
  for (int e = threadIdx.x; e < 97 * 97; e += 256) {
    int r = e / 97, c = e % 97;
    float s = 0.f;
    for (int cc = 0; cc < 32; ++cc) s += vs[r * 128 + c + cc];
    if (s > mx) { mx = s; mi = e; }
  }
  __shared__ float sM[256];
  __shared__ int sI[256];
  int t = threadIdx.x;
  sM[t] = mx; sI[t] = mi;
  __syncthreads();
  for (int off = 128; off > 0; off >>= 1) {
    if (t < off) {
      if (sM[t + off] > sM[t] || (sM[t + off] == sM[t] && sI[t + off] < sI[t])) {
        sM[t] = sM[t + off]; sI[t] = sI[t + off];
      }
    }
    __syncthreads();
  }
  if (t == 0) {
    float x2 = (sM[0] > 0.f) ? (float)(sI[0] / 97 + 16) : 0.f;
    float y2 = (sM[0] > 0.f) ? (float)(sI[0] % 97 + 16) : 0.f;
    scal[16 + b] = x2; scal[24 + b] = y2;
  }
}

// ---------------- K4: affine translation grids ----------------
__global__ void ma_grids_kernel(const float* __restrict__ scal,
                                float* __restrict__ g1, float* __restrict__ g2) {
  int i = blockIdx.x * 256 + threadIdx.x;
  if (i >= Bn * Kk) return;
  int b = i >> 14, p = i & (Kk - 1);
  int ii = p >> 7, jj = p & 127;
  float tx = (scal[16 + b] - scal[b]) * (1.f / 64.f);
  float ty = (scal[24 + b] - scal[8 + b]) * (1.f / 64.f);
  float xs = (2.f * jj + 1.f) / 128.f - 1.f;
  float ys = (2.f * ii + 1.f) / 128.f - 1.f;
  size_t o = (size_t)i * 2;
  g2[o] = xs + tx; g2[o + 1] = ys + ty;
  g1[o] = xs - tx; g1[o + 1] = ys - ty;
}

// ---------------- K5a: channel sum of f_g ----------------
__global__ void ma_fsum_kernel(const float* __restrict__ fg, float* __restrict__ fsum) {
  int i = blockIdx.x * 256 + threadIdx.x;
  if (i >= Bn * 65536) return;
  int b = i >> 16, p = i & 65535;
  const float* s = fg + (size_t)b * 3 * 65536;
  fsum[i] = s[p] + s[65536 + p] + s[131072 + p];
}

// ---------------- K5b: Gaussian-derivative edge @256 ----------------
__global__ void ma_edge256_kernel(const float* __restrict__ fsum, float* __restrict__ edge) {
  int i = blockIdx.x * 256 + threadIdx.x;
  if (i >= Bn * 65536) return;
  int b = i >> 16, p = i & 65535;
  int y = p >> 8, x = p & 255;
  const float* s = fsum + (size_t)b * 65536;
  float h = 0.f, v = 0.f;
  #pragma unroll
  for (int r = 0; r < 5; ++r) {
    #pragma unroll
    for (int c = 0; c < 5; ++c) {
      int yy = y + r - 2, xx = x + c - 2;
      if (yy < 0 || yy >= 256 || xx < 0 || xx >= 256) continue;
      float axr = (float)(r - 2), axc = (float)(c - 2);
      float g = __expf(-(axc * axc + axr * axr) * 0.5f) * 0.15915494309189535f;
      float val = s[yy * 256 + xx];
      h += val * (-axc * g);
      v += val * (-axr * g);
    }
  }
  edge[i] = tanhf(sqrtf(h * h + v * v + 1e-8f));
}

// ---------------- generic antialiased 2x downsample (jax.image.resize bilinear) -----
__global__ void ma_down2x_kernel(const float* __restrict__ src, float* __restrict__ dst,
                                 int total) {
  int i = blockIdx.x * 256 + threadIdx.x;
  if (i >= total) return;
  int map = i >> 14, p = i & (Kk - 1);
  int oy = p >> 7, ox = p & 127;
  const float* s = src + (size_t)map * 65536;
  const float bw[4] = {1.f, 3.f, 3.f, 1.f};
  float wx_sum = 0.f, wy_sum = 0.f, acc = 0.f;
  #pragma unroll
  for (int t = 0; t < 4; ++t) {
    int j = 2 * ox - 1 + t;
    if (j >= 0 && j < 256) wx_sum += bw[t];
  }
  #pragma unroll
  for (int ty = 0; ty < 4; ++ty) {
    int jy = 2 * oy - 1 + ty;
    if (jy < 0 || jy >= 256) continue;
    wy_sum += bw[ty];
    float rowacc = 0.f;
    #pragma unroll
    for (int tx = 0; tx < 4; ++tx) {
      int jx = 2 * ox - 1 + tx;
      if (jx < 0 || jx >= 256) continue;
      rowacc += bw[tx] * s[jy * 256 + jx];
    }
    acc += bw[ty] * rowacc;
  }
  dst[i] = acc / (wy_sum * wx_sum);
}

// ---------------- K7: conv1 (3ch identical edge -> summed weights) + relu6 ----------
__global__ void ma_conv1_kernel(const float* __restrict__ e128, const float* __restrict__ w,
                                const float* __restrict__ bias, float* __restrict__ out) {
  int i = blockIdx.x * 256 + threadIdx.x;
  if (i >= (int)FEAT) return;
  int b = i >> 20, o = (i >> 14) & 63, p = i & (Kk - 1);
  int y = p >> 7, x = p & 127;
  const float* s = e128 + (size_t)b * Kk;
  float acc = bias[o];
  #pragma unroll
  for (int u = 0; u < 3; ++u) {
    #pragma unroll
    for (int v = 0; v < 3; ++v) {
      int yy = y + u - 1, xx = x + v - 1;
      if (yy < 0 || yy >= 128 || xx < 0 || xx >= 128) continue;
      float we = w[((o * 3 + 0) * 3 + u) * 3 + v] + w[((o * 3 + 1) * 3 + u) * 3 + v]
               + w[((o * 3 + 2) * 3 + u) * 3 + v];
      acc += s[yy * 128 + xx] * we;
    }
  }
  out[i] = fminf(fmaxf(acc, 0.f), 6.f);
}

// ---------------- K8: 5x5 color moments @256 ----------------
__global__ void ma_mom256_kernel(const float* __restrict__ fe, float* __restrict__ mom) {
  int i = blockIdx.x * 256 + threadIdx.x;
  if (i >= Bn * 3 * 65536) return;
  int b = i / (3 * 65536), rem = i % (3 * 65536);
  int c = rem >> 16, p = rem & 65535;
  int y = p >> 8, x = p & 255;
  const float* s = fe + ((size_t)b * 3 + c) * 65536;
  float s1 = 0.f, s2 = 0.f, s3 = 0.f;
  #pragma unroll
  for (int dy = -2; dy <= 2; ++dy) {
    #pragma unroll
    for (int dx = -2; dx <= 2; ++dx) {
      int yy = y + dy, xx = x + dx;
      if (yy < 0 || yy >= 256 || xx < 0 || xx >= 256) continue;
      float v = s[yy * 256 + xx];
      float v2 = v * v;
      s1 += v; s2 += v2; s3 += v2 * v;
    }
  }
  const float inv_n = 1.f / 25.f;
  float mean = s1 * inv_n;
  float m2 = s2 * inv_n - mean * mean;
  float var = m2 * (25.f / 24.f);
  float m3 = s3 * inv_n - 3.f * mean * (s2 * inv_n) + 2.f * mean * mean * mean;
  float stdv = sqrtf(fmaxf(var, 1e-12f));
  float skew = m3 / (stdv * stdv * stdv + 1e-6f);
  size_t base = (size_t)b * 9 * 65536 + p;
  mom[base + (size_t)c * 65536]       = mean;
  mom[base + (size_t)(c + 3) * 65536] = var;
  mom[base + (size_t)(c + 6) * 65536] = skew;
}

// ---------------- K9: conv2 (9ch color) + bias ----------------
__global__ void ma_conv2_kernel(const float* __restrict__ col, const float* __restrict__ w,
                                const float* __restrict__ bias, float* __restrict__ out) {
  int i = blockIdx.x * 256 + threadIdx.x;
  if (i >= (int)FEAT) return;
  int b = i >> 20, o = (i >> 14) & 63, p = i & (Kk - 1);
  int y = p >> 7, x = p & 127;
  float acc = bias[o];
  for (int ci = 0; ci < 9; ++ci) {
    const float* s = col + ((size_t)b * 9 + ci) * Kk;
    #pragma unroll
    for (int u = 0; u < 3; ++u) {
      #pragma unroll
      for (int v = 0; v < 3; ++v) {
        int yy = y + u - 1, xx = x + v - 1;
        if (yy < 0 || yy >= 128 || xx < 0 || xx >= 128) continue;
        acc += s[yy * 128 + xx] * w[((o * 9 + ci) * 3 + u) * 3 + v];
      }
    }
  }
  out[i] = acc;
}

// ---------------- K10: masked translate (grid_sample) -> bf16 moves ----------------
__global__ void ma_move_kernel(const float* __restrict__ edgeF, const float* __restrict__ colorF,
                               const float* __restrict__ scal,
                               __bf16* __restrict__ f2m, __bf16* __restrict__ f1m) {
  size_t i = (size_t)blockIdx.x * 256 + threadIdx.x;
  if (i >= 2 * FEAT) return;
  int dir = (int)(i >> 23);
  size_t r = i & (FEAT - 1);
  int b = (int)(r >> 20), c = (int)((r >> 14) & 63), p = (int)(r & (Kk - 1));
  int ii = p >> 7, jj = p & 127;
  float x1 = scal[b], y1 = scal[8 + b], x2 = scal[16 + b], y2 = scal[24 + b];
  float cx, cy, dy, dx;
  const float* src;
  __bf16* dst;
  if (dir == 0) { src = colorF; dst = f2m; cx = x2; cy = y2; dy = y2 - y1; dx = x2 - x1; }
  else          { src = edgeF;  dst = f1m; cx = x1; cy = y1; dy = y1 - y2; dx = x1 - x2; }
  const float* s = src + ((size_t)b * 64 + c) * Kk;
  float sy = (float)ii + dy, sx = (float)jj + dx;
  float y0f = floorf(sy), x0f = floorf(sx);
  float fy = sy - y0f, fx = sx - x0f;
  int iy0 = (int)y0f, ix0 = (int)x0f;
  auto samp = [&](int yy, int xx) -> float {
    if (yy < 0 || yy >= 128 || xx < 0 || xx >= 128) return 0.f;
    float fyy = (float)yy, fxx = (float)xx;
    bool m = (fyy >= cx - 16.f) && (fyy < cx + 16.f) && (fxx >= cy - 16.f) && (fxx < cy + 16.f);
    return m ? s[yy * 128 + xx] : 0.f;
  };
  float acc = samp(iy0, ix0) * (1.f - fx) * (1.f - fy)
            + samp(iy0, ix0 + 1) * fx * (1.f - fy)
            + samp(iy0 + 1, ix0) * (1.f - fx) * fy
            + samp(iy0 + 1, ix0 + 1) * fx * fy;
  dst[r] = (__bf16)acc;
}

// ---------------- K11: score GEMM  S[c,d] = sum_k move[c,k]*feat[d,k]  (WMMA) -------
// grid: (8 k-chunks, 16 batch*dir); block 128 (4 waves = 4 m-strips)
__global__ void ma_score_kernel(const float* __restrict__ feat1, const float* __restrict__ feat2,
                                const __bf16* __restrict__ f2m, const __bf16* __restrict__ f1m,
                                float* __restrict__ part) {
  const int kchunk = blockIdx.x;
  const int bd = blockIdx.y;
  const int dir = bd >> 3, b = bd & 7;
  const __bf16* Amat = (dir == 0 ? f2m : f1m) + (size_t)b * Cc * Kk;
  const float*  Bmat = (dir == 0 ? feat1 : feat2) + (size_t)b * Cc * Kk;
  const int wave = threadIdx.x >> 5, lane = threadIdx.x & 31;
  const int lm = lane & 15, lh = lane >> 4;
  const int m0 = wave * 16;
  v8f acc0 = {}, acc1 = {}, acc2 = {}, acc3 = {};
  const int kbeg = kchunk * (Kk / 8), kend = kbeg + (Kk / 8);
  const __bf16* arow = Amat + (size_t)(m0 + lm) * Kk;
  for (int kb = kbeg; kb < kend; kb += 32) {
    v8bf alo = *(const v8bf*)(arow + kb + 16 * lh);
    v8bf ahi = *(const v8bf*)(arow + kb + 16 * lh + 8);
    v16bf av;
    #pragma unroll
    for (int t = 0; t < 8; ++t) { av[t] = alo[t]; av[8 + t] = ahi[t]; }
    #pragma unroll
    for (int j = 0; j < 4; ++j) {
      const float* brow = Bmat + (size_t)(j * 16 + lm) * Kk + kb + 16 * lh;
      v4f q0 = *(const v4f*)(brow);
      v4f q1 = *(const v4f*)(brow + 4);
      v4f q2 = *(const v4f*)(brow + 8);
      v4f q3 = *(const v4f*)(brow + 12);
      v16bf bv;
      #pragma unroll
      for (int t = 0; t < 4; ++t) {
        bv[t] = (__bf16)q0[t]; bv[4 + t] = (__bf16)q1[t];
        bv[8 + t] = (__bf16)q2[t]; bv[12 + t] = (__bf16)q3[t];
      }
      if (j == 0) acc0 = WMMA_BF16(av, bv, acc0);
      else if (j == 1) acc1 = WMMA_BF16(av, bv, acc1);
      else if (j == 2) acc2 = WMMA_BF16(av, bv, acc2);
      else acc3 = WMMA_BF16(av, bv, acc3);
    }
  }
  float* p = part + (size_t)(kchunk * 16 + bd) * 64 * 64;
  #pragma unroll
  for (int j = 0; j < 4; ++j) {
    v8f acc = (j == 0) ? acc0 : (j == 1) ? acc1 : (j == 2) ? acc2 : acc3;
    #pragma unroll
    for (int t = 0; t < 8; ++t)
      p[(m0 + 8 * lh + t) * 64 + j * 16 + lm] = acc[t];
  }
}

// ---------------- K12: sum partials + row softmax -> attn f32 out + bf16 ----------
__global__ void ma_softmax_kernel(const float* __restrict__ part,
                                  float* __restrict__ out7, float* __restrict__ out8,
                                  __bf16* __restrict__ attnbf) {
  __shared__ float S[64 * 64];
  const int bd = blockIdx.x;
  const int dir = bd >> 3, b = bd & 7;
  const int c = threadIdx.x;
  for (int d = 0; d < 64; ++d) {
    float s = 0.f;
    for (int ch = 0; ch < 8; ++ch)
      s += part[((size_t)(ch * 16 + bd) * 64 + c) * 64 + d];
    S[c * 64 + d] = s;
  }
  float mx = -1e30f;
  for (int d = 0; d < 64; ++d) mx = fmaxf(mx, S[c * 64 + d]);
  float sum = 0.f;
  for (int d = 0; d < 64; ++d) {
    float e = __expf(S[c * 64 + d] - mx);
    S[c * 64 + d] = e; sum += e;
  }
  float inv = 1.f / sum;
  float* dstf = ((dir == 1) ? out7 : out8) + ((size_t)b * 64 + c) * 64;
  __bf16* dstb = attnbf + (size_t)bd * 4096 + c * 64;
  for (int d = 0; d < 64; ++d) {
    float a = S[c * 64 + d] * inv;
    dstf[d] = a;
    dstb[d] = (__bf16)a;
  }
}

// ---------------- K13: apply GEMM  D = attn(64x64) @ feat(64x16384)  (WMMA) --------
// grid: (64 n-chunks of 256, 16 batch*dir); block 128
__global__ void ma_apply_kernel(const float* __restrict__ feat1, const float* __restrict__ feat2,
                                const __bf16* __restrict__ attnbf, float* __restrict__ out) {
  const int nchunk = blockIdx.x;
  const int bd = blockIdx.y;
  const int dir = bd >> 3, b = bd & 7;
  const __bf16* A = attnbf + (size_t)bd * 4096;
  const float* Bmat = (dir == 0 ? feat1 : feat2) + (size_t)b * Cc * Kk;
  float* O = out + (size_t)dir * FEAT + (size_t)b * Cc * Kk;
  const int wave = threadIdx.x >> 5, lane = threadIdx.x & 31;
  const int lm = lane & 15, lh = lane >> 4;
  const int m0 = wave * 16;
  const __bf16* arow = A + (m0 + lm) * 64;
  v16bf a0, a1;
  {
    v8bf l0 = *(const v8bf*)(arow + 8 * lh);
    v8bf h0 = *(const v8bf*)(arow + 16 + 8 * lh);
    v8bf l1 = *(const v8bf*)(arow + 32 + 8 * lh);
    v8bf h1 = *(const v8bf*)(arow + 48 + 8 * lh);
    #pragma unroll
    for (int t = 0; t < 8; ++t) {
      a0[t] = l0[t]; a0[8 + t] = h0[t];
      a1[t] = l1[t]; a1[8 + t] = h1[t];
    }
  }
  const int nbase = nchunk * 256;
  for (int j = 0; j < 16; ++j) {
    const int n0 = nbase + j * 16;
    v8f acc = {};
    v16bf bv;
    #pragma unroll
    for (int t = 0; t < 16; ++t)
      bv[t] = (__bf16)Bmat[(size_t)(8 * lh + t + (t & 8)) * Kk + n0 + lm];
    acc = WMMA_BF16(a0, bv, acc);
    #pragma unroll
    for (int t = 0; t < 16; ++t)
      bv[t] = (__bf16)Bmat[(size_t)(32 + 8 * lh + t + (t & 8)) * Kk + n0 + lm];
    acc = WMMA_BF16(a1, bv, acc);
    #pragma unroll
    for (int t = 0; t < 8; ++t)
      O[(size_t)(m0 + 8 * lh + t) * Kk + n0 + lm] = acc[t];
  }
}

// ---------------- K14: BN stats per (dir, channel) ----------------
__global__ void ma_bnstat_kernel(const float* __restrict__ out, float* __restrict__ stat) {
  const int idx = blockIdx.x;            // 0..127
  const int dir = idx >> 6, ch = idx & 63;
  const float* p = out + (size_t)dir * FEAT;
  float s = 0.f, s2 = 0.f;
  for (int l = threadIdx.x; l < Bn * Kk; l += 256) {
    int bb = l >> 14, sp = l & (Kk - 1);
    float v = p[((size_t)bb * 64 + ch) * Kk + sp];
    s += v; s2 += v * v;
  }
  __shared__ float sh[256], sh2[256];
  int t = threadIdx.x;
  sh[t] = s; sh2[t] = s2;
  __syncthreads();
  for (int off = 128; off > 0; off >>= 1) {
    if (t < off) { sh[t] += sh[t + off]; sh2[t] += sh2[t + off]; }
    __syncthreads();
  }
  if (t == 0) {
    const float invN = 1.f / (float)(Bn * Kk);
    float mean = sh[0] * invN;
    float var = sh2[0] * invN - mean * mean;   // biased, matches jnp.var
    stat[idx * 2] = mean;
    stat[idx * 2 + 1] = rsqrtf(var + 1e-5f);
  }
}

// ---------------- K15: BN apply (in place) + relu6 + warm_flag select --------------
__global__ void ma_bnapply_kernel(float* __restrict__ out, const float* __restrict__ stat,
                                  const int* __restrict__ warm,
                                  const float* __restrict__ feat1,
                                  const float* __restrict__ feat2) {
  size_t i = (size_t)blockIdx.x * 256 + threadIdx.x;
  if (i >= 2 * FEAT) return;
  int dir = (int)(i >> 23);
  size_t r = i & (FEAT - 1);
  int ch = (int)((r >> 14) & 63);
  float mean = stat[(dir * 64 + ch) * 2];
  float rstd = stat[(dir * 64 + ch) * 2 + 1];
  float y = fminf(fmaxf((out[i] - mean) * rstd, 0.f), 6.f);
  if (*warm == 0) y = ((dir == 0) ? feat1 : feat2)[r];
  out[i] = y;
}

// ---------------- launcher ----------------
extern "C" void kernel_launch(void* const* d_in, const int* in_sizes, int n_in,
                              void* d_out, int out_size, void* d_ws, size_t ws_size,
                              hipStream_t stream) {
  (void)in_sizes; (void)n_in; (void)out_size; (void)ws_size;
  const int*   warm    = (const int*)d_in[0];
  const float* feat1   = (const float*)d_in[1];
  const float* feat2   = (const float*)d_in[2];
  const float* cam1    = (const float*)d_in[3];
  const float* cam2    = (const float*)d_in[4];
  const float* f_g     = (const float*)d_in[5];
  const float* f_e     = (const float*)d_in[6];
  const float* conv1_w = (const float*)d_in[7];
  const float* conv1_b = (const float*)d_in[8];
  const float* conv2_w = (const float*)d_in[9];
  const float* conv2_b = (const float*)d_in[10];
  float* out = (float*)d_out;
  char* ws = (char*)d_ws;

  float* c1     = (float*)(ws + WS_C1);
  float* c2     = (float*)(ws + WS_C2);
  float* vsum   = (float*)(ws + WS_VSUM);
  float* scal   = (float*)(ws + WS_SCAL);
  float* fsum   = (float*)(ws + WS_FSUM);
  float* e256   = (float*)(ws + WS_E256);
  float* e128   = (float*)(ws + WS_E128);
  float* mom    = (float*)(ws + WS_MOM);
  float* col    = (float*)(ws + WS_COL);
  __bf16* f2m   = (__bf16*)(ws + WS_F2M);
  __bf16* f1m   = (__bf16*)(ws + WS_F1M);
  float* part   = (float*)(ws + WS_PART);
  __bf16* attnb = (__bf16*)(ws + WS_ATTN);
  float* bnstat = (float*)(ws + WS_BN);

  // passthrough outputs 9/10
  hipMemcpyAsync(out + O9, feat1, FEAT * sizeof(float), hipMemcpyDeviceToDevice, stream);
  hipMemcpyAsync(out + O10, feat2, FEAT * sizeof(float), hipMemcpyDeviceToDevice, stream);

  // cams, centers, grids
  ma_up2x_kernel<<<512, 256, 0, stream>>>(cam1, c1);
  ma_up2x_kernel<<<512, 256, 0, stream>>>(cam2, c2);
  ma_stats1_kernel<<<8, 256, 0, stream>>>(c1, scal);
  ma_vsum_kernel<<<(8 * 97 * 128 + 255) / 256, 256, 0, stream>>>(c2, vsum);
  ma_stats2_kernel<<<8, 256, 0, stream>>>(vsum, scal);
  ma_grids_kernel<<<512, 256, 0, stream>>>(scal, out + O3, out + O4);

  // edge stream -> feat1_edge (slot 5)
  ma_fsum_kernel<<<2048, 256, 0, stream>>>(f_g, fsum);
  ma_edge256_kernel<<<2048, 256, 0, stream>>>(fsum, e256);
  ma_down2x_kernel<<<512, 256, 0, stream>>>(e256, e128, 8 * Kk);
  ma_conv1_kernel<<<32768, 256, 0, stream>>>(e128, conv1_w, conv1_b, out + O5);

  // color stream -> feat2_color (slot 6)
  ma_mom256_kernel<<<6144, 256, 0, stream>>>(f_e, mom);
  ma_down2x_kernel<<<4608, 256, 0, stream>>>(mom, col, 72 * Kk);
  ma_conv2_kernel<<<32768, 256, 0, stream>>>(col, conv2_w, conv2_b, out + O6);

  // masked translate -> bf16 moves
  ma_move_kernel<<<65536, 256, 0, stream>>>(out + O5, out + O6, scal, f2m, f1m);

  // attention GEMMs + softmax
  ma_score_kernel<<<dim3(8, 16), 128, 0, stream>>>(feat1, feat2, f2m, f1m, part);
  ma_softmax_kernel<<<16, 64, 0, stream>>>(part, out + O7, out + O8, attnb);
  ma_apply_kernel<<<dim3(64, 16), 128, 0, stream>>>(feat1, feat2, attnb, out);

  // batch-norm (training stats) + relu6, in place on slots 1/2
  ma_bnstat_kernel<<<128, 256, 0, stream>>>(out, bnstat);
  ma_bnapply_kernel<<<65536, 256, 0, stream>>>(out, bnstat, warm, feat1, feat2);
}
